// TopDownLSTMEncoder_75797582839903
// MI455X (gfx1250) — matrix operand order repaired and checked by reading, not compiled, not run on hardware
//
#include <hip/hip_runtime.h>
#include <hip/hip_bf16.h>

// ---------------------------------------------------------------------------
// Tree-LSTM (top-down), B=32, N=1024, D=256, H=256 on MI455X (gfx1250).
//
//  prep_U:      U (512 x 768) bf16, pre-swizzled into WMMA B-fragment layout.
//                 rows   0..255 : V = [tf^T tf | to^T to | tz^T tz]
//                 rows 256..511 : W_inp
//  prep_gates:  gatesE = emb @ W_inp + b  (one-time 12.9-GFLOP WMMA GEMM,
//               2048 blocks), stored bf16 (B*N, 768).
//  scan:        2 blocks x 512 thr (16 waves, 16 batches each, M=16).
//               Recurrent weights V (3 n-tiles x 8 k-chunks x 8 VGPRs =
//               192 VGPRs/wave) are preloaded into REGISTERS and stay
//               resident for all 1024 steps -> zero weight traffic on the
//               sequential critical path. Per step: direct per-lane A-frag
//               gather from a bf16 h-mirror, 24 v_wmma, LDS gate combine,
//               pointwise LSTM with operand prefetch hidden under the
//               gate barrier.
// ---------------------------------------------------------------------------

#define NNODES  1024
#define HDIM    256
#define NGATE   768
#define KT      16           // k-chunks in the full U (512 rows / 32)
#define KTV     8            // k-chunks of the recurrent V part (256 / 32)
#define NT      48           // n-tiles (768 / 16)
#define THREADS 512
#define TPW     3            // n-tiles per wave in scan (48 / 16 waves)
#define GST     776          // LDS stride (f32) for gate buffer
#define EST     272          // LDS stride (bf16) for emb staging in prep_gates

typedef __attribute__((ext_vector_type(16))) __bf16 v16bf;
typedef __attribute__((ext_vector_type(8)))  float  v8f;

struct alignas(16) U128 { unsigned int x[4]; };
union Frag16 { v16bf v; U128 q[2]; };

static __device__ __forceinline__ unsigned short f2bf(float x) {
    unsigned u = __float_as_uint(x);
    u += 0x7FFFu + ((u >> 16) & 1u);              // round-to-nearest-even
    return (unsigned short)(u >> 16);
}
static __device__ __forceinline__ float bf2f(unsigned short b) {
    return __uint_as_float((unsigned)b << 16);
}
static __device__ __forceinline__ float sigmoidf(float x) {
    return 1.0f / (1.0f + __expf(-x));
}

// ---------------------------------------------------------------------------
// U[k][n] (512 x 768) swizzled: frag(kt,nt) at ((nt*KT+kt)*512) bf16;
// within a frag: lane = ((k%32)/16)*16 + n%16, slot = k%16.
// ---------------------------------------------------------------------------
__global__ __launch_bounds__(256)
void prep_U(const float* __restrict__ W, const float* __restrict__ tf,
            const float* __restrict__ to, const float* __restrict__ tz,
            unsigned short* __restrict__ U) {
    int idx = blockIdx.x * blockDim.x + threadIdx.x;    // 0 .. 512*768-1
    if (idx >= 512 * NGATE) return;
    int krow = idx / NGATE;
    int ncol = idx - krow * NGATE;
    float val;
    if (krow < HDIM) {
        const float* T = (ncol < 256) ? tf : (ncol < 512 ? to : tz);
        int n2 = ncol & 255;
        float s = 0.f;
        #pragma unroll 4
        for (int d = 0; d < HDIM; ++d)
            s += T[d * HDIM + krow] * T[d * HDIM + n2]; // (T^T T)[k][n2]
        val = s;
    } else {
        val = W[(size_t)(krow - HDIM) * NGATE + ncol];
    }
    int kt = krow >> 5, kin = krow & 31;
    int nt = ncol >> 4, nin = ncol & 15;
    int lane = ((kin >> 4) << 4) | nin;
    int slot = kin & 15;
    U[((size_t)(nt * KT + kt) << 9) + (lane << 4) + slot] = f2bf(val);
}

// ---------------------------------------------------------------------------
// gatesE(row, col) = emb(row,:) @ W(:,col) + bias[col], bf16.  One block per
// 16-row M-tile (2048 blocks x 256 thr / 8 waves, 6 n-tiles per wave).
// ---------------------------------------------------------------------------
__global__ __launch_bounds__(256)
void prep_gates(const float* __restrict__ emb, const unsigned short* __restrict__ U,
                const float* __restrict__ bias, unsigned short* __restrict__ gatesE) {
    __shared__ alignas(16) unsigned short lsE[16 * EST];
    const int tid  = threadIdx.x;
    const int lane = tid & 31;
    const int wave = tid >> 5;
    const int m    = lane & 15;
    const int hi   = lane >> 4;
    const int r0   = blockIdx.x * 16;                   // flat (b*N+n) row base

    for (int idx = tid; idx < 16 * HDIM; idx += 256) {
        int r = idx >> 8, k = idx & 255;
        lsE[r * EST + k] = f2bf(emb[(size_t)(r0 + r) * HDIM + k]);
    }
    __syncthreads();

    v8f acc[6];
    #pragma unroll
    for (int t = 0; t < 6; ++t)
        #pragma unroll
        for (int j = 0; j < 8; ++j) acc[t][j] = 0.f;

    for (int kc = 0; kc < 8; ++kc) {                    // W part: kt = 8+kc
        Frag16 a;
        int kb = (kc << 5) + (hi << 4);
        a.q[0] = *(const U128*)&lsE[m * EST + kb];
        a.q[1] = *(const U128*)&lsE[m * EST + kb + 8];
        #pragma unroll
        for (int t = 0; t < 6; ++t) {
            int nt = wave * 6 + t;
            Frag16 b;
            const U128* fp = (const U128*)&U[((size_t)(nt * KT + 8 + kc) << 9) + (lane << 4)];
            b.q[0] = fp[0];  b.q[1] = fp[1];
            acc[t] = __builtin_amdgcn_wmma_f32_16x16x32_bf16(
                false, a.v, false, b.v, (short)0, acc[t], false, false);
        }
    }
    #pragma unroll
    for (int t = 0; t < 6; ++t) {
        int col = (wave * 6 + t) * 16 + m;
        float bcol = bias[col];
        #pragma unroll
        for (int r = 0; r < 8; ++r) {
            int row = r0 + hi * 8 + r;
            gatesE[(size_t)row * NGATE + col] = f2bf(acc[t][r] + bcol);
        }
    }
}

// ---------------------------------------------------------------------------
// Sequential scan.  blockIdx.x in {0,1}: batches [blk*16, blk*16+16).
// ---------------------------------------------------------------------------
__global__ __launch_bounds__(THREADS)
void scan_kernel(const int* __restrict__ conn,
                 const unsigned short* __restrict__ U,
                 const unsigned short* __restrict__ gatesE,
                 unsigned short* __restrict__ hmir,        // bf16 h history
                 float* __restrict__ hs,                   // (B,N,H) = d_out
                 float* __restrict__ cs) {                 // (B,N,H) scratch
    __shared__ unsigned short lsConn[16 * NNODES];         // 32 KB parent ids
    __shared__ float lsG[16 * GST];                        // 48.5 KB gate sums

    const int tid  = threadIdx.x;
    const int lane = tid & 31;
    const int wave = tid >> 5;
    const int b0   = blockIdx.x * 16;
    const int m    = lane & 15;
    const int hi   = lane >> 4;

    // ---- recurrent weights V: preload to registers, resident for all steps
    Frag16 Bres[TPW][KTV];
    #pragma unroll
    for (int t = 0; t < TPW; ++t)
        #pragma unroll
        for (int kc = 0; kc < KTV; ++kc) {
            const U128* fp =
                (const U128*)&U[((size_t)((wave * TPW + t) * KT + kc) << 9) + (lane << 4)];
            Bres[t][kc].q[0] = fp[0];
            Bres[t][kc].q[1] = fp[1];
        }

    // ---- stage this block's parent-id table into LDS once
    for (int idx = tid; idx < 16 * NNODES; idx += THREADS) {
        int r = idx >> 10, n = idx & (NNODES - 1);
        lsConn[idx] = (unsigned short)conn[(size_t)(b0 + r) * NNODES + n];
    }
    __syncthreads();

    for (int i = 0; i < NNODES; ++i) {
        if (i > 0) {
            // ---- G(16x768) = ph(16x256) @ V ; A gathered per lane from hmir
            int pid = (int)lsConn[m * NNODES + i];
            const unsigned short* arow =
                hmir + (((size_t)(b0 + m)) * NNODES + pid) * HDIM;

            v8f acc[TPW];
            #pragma unroll
            for (int t = 0; t < TPW; ++t)
                #pragma unroll
                for (int j = 0; j < 8; ++j) acc[t][j] = 0.f;

            #pragma unroll
            for (int kc = 0; kc < KTV; ++kc) {
                Frag16 a;
                int kb = (kc << 5) + (hi << 4);
                a.q[0] = *(const U128*)&arow[kb];
                a.q[1] = *(const U128*)&arow[kb + 8];
                #pragma unroll
                for (int t = 0; t < TPW; ++t)
                    acc[t] = __builtin_amdgcn_wmma_f32_16x16x32_bf16(
                        false, a.v, false, Bres[t][kc].v, (short)0, acc[t],
                        false, false);
            }
            #pragma unroll
            for (int t = 0; t < TPW; ++t) {
                int nc = (wave * TPW + t) * 16 + m;
                int mb = hi * 8;
                #pragma unroll
                for (int r = 0; r < 8; ++r)
                    lsG[(mb + r) * GST + nc] = acc[t][r];
            }
        }

        // ---- prefetch pointwise operands (latency hidden under gate barrier)
        float pgf[8], pgo[8], pgz[8], ppc[8];
        #pragma unroll
        for (int j = 0; j < 8; ++j) {
            int e = tid + j * THREADS;                 // 0 .. 16*256-1
            int br = e >> 8, h = e & 255;
            size_t grow = (((size_t)(b0 + br)) * NNODES + i) * NGATE;
            pgf[j] = bf2f(gatesE[grow + h]);
            pgo[j] = bf2f(gatesE[grow + 256 + h]);
            pgz[j] = bf2f(gatesE[grow + 512 + h]);
            if (i > 0) {
                int pid = (int)lsConn[br * NNODES + i];
                ppc[j] = cs[(((size_t)(b0 + br)) * NNODES + pid) * HDIM + h];
            } else {
                ppc[j] = 0.f;
            }
        }
        __syncthreads();

        // ---- pointwise LSTM cell
        #pragma unroll
        for (int j = 0; j < 8; ++j) {
            int e = tid + j * THREADS;
            int br = e >> 8, h = e & 255;
            float gf = pgf[j], go = pgo[j], gz = pgz[j];
            if (i > 0) {
                gf += lsG[br * GST + h];
                go += lsG[br * GST + 256 + h];
                gz += lsG[br * GST + 512 + h];
            }
            float f = sigmoidf(gf);
            float o = sigmoidf(go);
            float z = tanhf(gz);
            float c = ppc[j] * f + z * (1.f - f);
            float hv = o * tanhf(c);
            size_t oidx = (((size_t)(b0 + br)) * NNODES + i) * HDIM + h;
            cs[oidx]   = c;
            hs[oidx]   = hv;
            hmir[oidx] = f2bf(hv);
        }
        __threadfence();       // publish h/c history before next step's gather
        __syncthreads();
    }
}

extern "C" void kernel_launch(void* const* d_in, const int* in_sizes, int n_in,
                              void* d_out, int out_size, void* d_ws, size_t ws_size,
                              hipStream_t stream) {
    const float* emb  = (const float*)d_in[0];  // (B,N,D)
    const int*   conn = (const int*)  d_in[1];  // (B,N)
    // d_in[2] node_mask: unused by reference math
    const float* W    = (const float*)d_in[3];  // (D, 3H)
    const float* bias = (const float*)d_in[4];  // (3H,)
    const float* tf   = (const float*)d_in[5];  // (H,H)
    const float* to   = (const float*)d_in[6];
    const float* tz   = (const float*)d_in[7];

    char* ws = (char*)d_ws;
    unsigned short* U      = (unsigned short*)ws;                         // 768 KB
    unsigned short* gatesE = (unsigned short*)(ws + 786432);              // 48 MB
    unsigned short* hmir   = (unsigned short*)(ws + 786432 + 50331648);   // 16 MB
    float*          cs     = (float*)(ws + 786432 + 50331648 + 16777216); // 32 MB
    float*          out    = (float*)d_out;

    prep_U<<<(512 * NGATE + 255) / 256, 256, 0, stream>>>(W, tf, to, tz, U);
    prep_gates<<<(32 * NNODES) / 16, 256, 0, stream>>>(emb, U, bias, gatesE);
    scan_kernel<<<2, THREADS, 0, stream>>>(conn, U, gatesE, hmir, out, cs);
}
